// GNNPolicy_57638461112993
// MI455X (gfx1250) — matrix operand order repaired and checked by reading, not compiled, not run on hardware
//
#include <hip/hip_runtime.h>

// ---------------------------------------------------------------------------
// GNN policy forward for MI455X (gfx1250, wave32, WMMA bf16 16x16x32).
// All linear layers run on v_wmma_f32_16x16x32_bf16 with bf16 weights
// pre-converted once (L2-resident); 4 WMMAs per A-fragment per wave.
// GEMMs are templated on A/C precision mode -> branch-free staging.
// ---------------------------------------------------------------------------

#define Bc 4
#define Nn 256
#define Ee 128
#define Hh 8
#define NE 1024          // Bc*Nn
#define NT 1025          // NE + 1 pad node
#define Gg 5             // Bc + 1 pad graph
#define ESIZE 130560     // Bc*Nn*(Nn-1)/2  (multiple of 64)

typedef __attribute__((ext_vector_type(16))) __bf16 v16bf;
typedef __attribute__((ext_vector_type(8)))  float  v8f;

union FragAB { v16bf v; unsigned int u[8]; };

__device__ __forceinline__ unsigned short f2bf(float f) {
  unsigned int u = __float_as_uint(f);
  u += 0x7FFFu + ((u >> 16) & 1u);     // round-to-nearest-even
  return (unsigned short)(u >> 16);
}
__device__ __forceinline__ unsigned int f2bf2(float a, float b) {
  return (unsigned int)f2bf(a) | ((unsigned int)f2bf(b) << 16);
}

// fp32 -> packed bf16 pair converter (weights, rc matrix)
__global__ void w2bf_kernel(const float* __restrict__ W, unsigned int* __restrict__ Wb, int n2) {
  int i = blockIdx.x * 256 + threadIdx.x;
  if (i < n2) Wb[i] = f2bf2(W[2 * i], W[2 * i + 1]);
}

// ---------------------------------------------------------------------------
// GEMM:  C[M, Nout] = A[:, acol:acol+K] @ W[Nout,K]^T (+bias) (opt ReLU)
// ABF: A is packed bf16 (else fp32).  OBF: output bf16 (else fp32).
// W is packed bf16 pairs, ldw2 = K/2 uints per row.
// Block = 256 thr = 8 waves; tile 64(M) x 128(N): wave = (row-tile, col-group),
// each wave: 1 A-frag -> 4 WMMAs per K-step.
// grid.x = ceil(M/64), grid.y = Nout/128.
// ---------------------------------------------------------------------------
template <bool ABF, bool OBF>
__launch_bounds__(256)
__global__ void gemm_t(const float* Af, const unsigned short* Ab,
                       int lda, int acol, int Mreal, int K,
                       const unsigned int* __restrict__ Wb, int ldw2,
                       const float* __restrict__ bias,
                       float* Cf, unsigned short* Cb,
                       int ldc, int relu) {
  __shared__ unsigned int As[64][16];        // 64x32 bf16 tile, packed pairs
  const int tid  = threadIdx.x;
  const int wave = tid >> 5, lane = tid & 31;
  const int rt = wave & 3;                   // row tile within block
  const int cg = wave >> 2;                  // column group (64 cols)
  const int mbase  = blockIdx.x * 64;
  const int nblock = blockIdx.y * 128;
  const int lh = lane >> 4;                  // lane half
  const int r  = lane & 15;
  v8f acc[4] = {{}, {}, {}, {}};
  for (int kb = 0; kb < K; kb += 32) {
    __syncthreads();
#pragma unroll
    for (int j = 0; j < 4; ++j) {            // stage A tile (branch-free mode)
      int e = tid + 256 * j;
      int row = e >> 4, cp = e & 15;
      int m = mbase + row;
      unsigned int pv = 0u;
      if (m < Mreal) {
        if constexpr (ABF) {
          pv = *(const unsigned int*)(Ab + (size_t)m * lda + acol + kb + 2 * cp);
        } else {
          const float* src = Af + (size_t)m * lda + acol + kb + 2 * cp;
          pv = f2bf2(src[0], src[1]);
        }
      }
      As[row][cp] = pv;
    }
    __syncthreads();
    FragAB a;
    const int kc = lh * 8;                   // halves hold K 0-15 vs 16-31
    const int ar = rt * 16 + r;
#pragma unroll
    for (int i = 0; i < 4; ++i) { a.u[i] = As[ar][kc + i]; a.u[4 + i] = As[ar][4 + kc + i]; }
    const int kw = (kb >> 1) + lh * 8;       // uint offset in W row
#pragma unroll
    for (int j = 0; j < 4; ++j) {
      FragAB b;
      const int n = nblock + cg * 64 + j * 16 + r;
      const unsigned int* wrow = Wb + (size_t)n * ldw2 + kw;
      __builtin_prefetch((const void*)(wrow + 16), 0, 1);
#pragma unroll
      for (int i = 0; i < 8; ++i) b.u[i] = wrow[i];
      acc[j] = __builtin_amdgcn_wmma_f32_16x16x32_bf16(false, a.v, false, b.v,
                                                       (short)0, acc[j], false, false);
    }
  }
#pragma unroll
  for (int j = 0; j < 4; ++j) {
    const int n = nblock + cg * 64 + j * 16 + r;
    const float bv = bias ? bias[n] : 0.f;
#pragma unroll
    for (int i = 0; i < 8; ++i) {
      int m = mbase + rt * 16 + i + lh * 8;  // C: VGPR i -> rows i / i+8
      if (m < Mreal) {
        float v = acc[j][i] + bv;
        if (relu) v = fmaxf(v, 0.f);
        if constexpr (OBF) Cb[(size_t)m * ldc + n] = f2bf(v);
        else               Cf[(size_t)m * ldc + n] = v;
      }
    }
  }
}

// ---------------------------------------------------------------------------
// Gather-concat GEMM: A row m = concat of NSEG fp32 sources (each 128 wide),
// source s optionally indexed via Is[m]. K = NSEG*128 (compile-time),
// ldw2 = NSEG*64, ldc = 128, output bf16.
// ---------------------------------------------------------------------------
template <int NSEG>
__launch_bounds__(256)
__global__ void gemm_cat_t(const float* __restrict__ S0, const float* __restrict__ S1,
                           const float* __restrict__ S2, const float* __restrict__ S3,
                           const int* __restrict__ I1, const int* __restrict__ I2,
                           const int* __restrict__ I3,
                           int Mreal,
                           const unsigned int* __restrict__ Wb,
                           const float* __restrict__ bias,
                           unsigned short* Cb, int relu) {
  __shared__ unsigned int As[64][16];
  const int tid  = threadIdx.x;
  const int wave = tid >> 5, lane = tid & 31;
  const int rt = wave & 3;
  const int cg = wave >> 2;
  const int mbase  = blockIdx.x * 64;
  const int nblock = blockIdx.y * 128;
  const int lh = lane >> 4;
  const int r  = lane & 15;
  constexpr int K = NSEG * 128;
  constexpr int ldw2 = NSEG * 64;
  v8f acc[4] = {{}, {}, {}, {}};
  for (int kb = 0; kb < K; kb += 32) {
    __syncthreads();
#pragma unroll
    for (int j = 0; j < 4; ++j) {
      int e = tid + 256 * j;
      int row = e >> 4, cp = e & 15;
      int m = mbase + row;
      int col = kb + 2 * cp;
      int seg = col >> 7, cs = col & 127;
      unsigned int pv = 0u;
      if (m < Mreal) {
        const float* base; int ri = m;
        if (seg == 0)      { base = S0; }
        else if (seg == 1) { base = S1; if (I1) ri = I1[m]; }
        else if (seg == 2) { base = S2; if (I2) ri = I2[m]; }
        else               { base = S3; if (I3) ri = I3[m]; }
        const float* src = base + (size_t)ri * 128 + cs;
        pv = f2bf2(src[0], src[1]);
      }
      As[row][cp] = pv;
    }
    __syncthreads();
    FragAB a;
    const int kc = lh * 8;
    const int ar = rt * 16 + r;
#pragma unroll
    for (int i = 0; i < 4; ++i) { a.u[i] = As[ar][kc + i]; a.u[4 + i] = As[ar][4 + kc + i]; }
    const int kw = (kb >> 1) + lh * 8;
#pragma unroll
    for (int j = 0; j < 4; ++j) {
      FragAB b;
      const int n = nblock + cg * 64 + j * 16 + r;
      const unsigned int* wrow = Wb + (size_t)n * ldw2 + kw;
      __builtin_prefetch((const void*)(wrow + 16), 0, 1);
#pragma unroll
      for (int i = 0; i < 8; ++i) b.u[i] = wrow[i];
      acc[j] = __builtin_amdgcn_wmma_f32_16x16x32_bf16(false, a.v, false, b.v,
                                                       (short)0, acc[j], false, false);
    }
  }
#pragma unroll
  for (int j = 0; j < 4; ++j) {
    const int n = nblock + cg * 64 + j * 16 + r;
    const float bv = bias ? bias[n] : 0.f;
#pragma unroll
    for (int i = 0; i < 8; ++i) {
      int m = mbase + rt * 16 + i + lh * 8;
      if (m < Mreal) {
        float v = acc[j][i] + bv;
        if (relu) v = fmaxf(v, 0.f);
        Cb[(size_t)m * 128 + n] = f2bf(v);
      }
    }
  }
}

// --------------------------- graph construction ----------------------------
__global__ void count_rows_kernel(const int* __restrict__ adj, int* __restrict__ counts) {
  __shared__ int red[256];
  int row = blockIdx.x, t = threadIdx.x;                 // row = g*Nn + s
  red[t] = (adj[(size_t)row * Nn + t] != 0) ? 1 : 0;
  __syncthreads();
  for (int s = 128; s > 0; s >>= 1) { if (t < s) red[t] += red[t + s]; __syncthreads(); }
  if (t == 0) counts[row] = red[0];
}

__global__ void scan_rows_kernel(const int* __restrict__ counts, int* __restrict__ offs) {
  __shared__ int buf[1024];
  int t = threadIdx.x;
  buf[t] = counts[t];
  __syncthreads();
  for (int d = 1; d < 1024; d <<= 1) {
    int v = (t >= d) ? buf[t - d] : 0;
    __syncthreads();
    buf[t] += v;
    __syncthreads();
  }
  offs[t] = buf[t] - counts[t];                          // exclusive
}

__global__ void init_edge_arrays_kernel(int* snd, int* rcv, int* egid) {
  int i = blockIdx.x * 256 + threadIdx.x;
  if (i < ESIZE) { snd[i] = NE; rcv[i] = NE; egid[i] = Bc; }   // padding values
}

__global__ void emit_edges_kernel(const int* __restrict__ adj, const int* __restrict__ offs,
                                  int* __restrict__ snd, int* __restrict__ rcv,
                                  int* __restrict__ egid) {
  int row = blockIdx.x * 256 + threadIdx.x;              // one thread per (g,s) row
  if (row >= NE) return;
  int g = row >> 8;
  int o = offs[row];
  for (int t = 0; t < Nn; ++t) {
    if (adj[(size_t)row * Nn + t] && o < ESIZE) {
      snd[o] = row; rcv[o] = g * Nn + t; egid[o] = g; ++o;
    }
  }
}

__global__ void init_ngid_kernel(int* ngid) {
  int i = blockIdx.x * 256 + threadIdx.x;
  if (i < NT) ngid[i] = (i < NE) ? (i >> 8) : Bc;
}

// ------------------------------ small helpers ------------------------------
__global__ void init_nodes_kernel(const float* __restrict__ node_emb, float* __restrict__ nodes) {
  int i = blockIdx.x, f = threadIdx.x;                   // NT blocks x 128
  int nid = (i < NE) ? (i & 255) : 0;
  nodes[(size_t)i * 128 + f] = node_emb[(size_t)nid * 128 + f];
}

__global__ void bcast_row_kernel(const float* __restrict__ src, float* __restrict__ dst) {
  dst[(size_t)blockIdx.x * 128 + threadIdx.x] = src[threadIdx.x];
}

__global__ void zero_kernel(float* p, int n) {
  int i = blockIdx.x * 256 + threadIdx.x;
  if (i < n) p[i] = 0.f;
}

__global__ void scatter_add_edges_kernel(const float* __restrict__ ne,
                                         const int* __restrict__ snd, const int* __restrict__ rcv,
                                         const int* __restrict__ egid,
                                         float* __restrict__ ssnd, float* __restrict__ srcv,
                                         float* __restrict__ seg) {
  int e = blockIdx.x, f = threadIdx.x;                   // ESIZE blocks x 128
  float v = ne[(size_t)e * 128 + f];
  atomicAdd(&ssnd[(size_t)snd[e]  * 128 + f], v);
  atomicAdd(&srcv[(size_t)rcv[e]  * 128 + f], v);
  atomicAdd(&seg [(size_t)egid[e] * 128 + f], v);
}

__global__ void scatter_add_nodes_kernel(const float* __restrict__ nn,
                                         const int* __restrict__ ngid,
                                         float* __restrict__ sng) {
  int i = blockIdx.x, f = threadIdx.x;                   // NT blocks x 128
  atomicAdd(&sng[(size_t)ngid[i] * 128 + f], nn[(size_t)i * 128 + f]);
}

// residual + layernorm: out = LN(x + delta); delta may be null
__global__ void ln_kernel(const float* __restrict__ x, const float* __restrict__ delta,
                          const float* __restrict__ w, const float* __restrict__ b,
                          float* __restrict__ out) {
  __shared__ float red[128];
  int row = blockIdx.x, f = threadIdx.x;
  size_t base = (size_t)row * 128;
  float v = x[base + f];
  if (delta) v += delta[base + f];
  red[f] = v; __syncthreads();
  for (int s = 64; s > 0; s >>= 1) { if (f < s) red[f] += red[f + s]; __syncthreads(); }
  float mu = red[0] * (1.0f / 128.0f);
  __syncthreads();
  float d = v - mu;
  red[f] = d * d; __syncthreads();
  for (int s = 64; s > 0; s >>= 1) { if (f < s) red[f] += red[f + s]; __syncthreads(); }
  float var = red[0] * (1.0f / 128.0f);
  out[base + f] = d * rsqrtf(var + 1e-5f) * w[f] + b[f];
}

// per-(b,h,q) softmax attention; grid = Bc*Hh*Nn blocks x 256 threads
__global__ void attn_kernel(const float* __restrict__ qh, const float* __restrict__ kh,
                            const float* __restrict__ vh, float* __restrict__ o) {
  __shared__ float qv[16];
  __shared__ float red[256];
  int bid = blockIdx.x;
  int qi = bid & 255, h = (bid >> 8) & 7, b = bid >> 11;
  int t = threadIdx.x;
  size_t rowq = ((size_t)(b * Nn + qi)) * 128 + h * 16;
  if (t < 16) qv[t] = qh[rowq + t];
  __syncthreads();
  size_t rowk = ((size_t)(b * Nn + t)) * 128 + h * 16;
  float s = 0.f;
#pragma unroll
  for (int d = 0; d < 16; ++d) s += qv[d] * kh[rowk + d];
  s *= 0.25f;                                            // 1/sqrt(dh), dh=16
  red[t] = s; __syncthreads();
  for (int st = 128; st > 0; st >>= 1) { if (t < st) red[t] = fmaxf(red[t], red[t + st]); __syncthreads(); }
  float mx = red[0]; __syncthreads();
  float p = __expf(s - mx);
  red[t] = p; __syncthreads();
  for (int st = 128; st > 0; st >>= 1) { if (t < st) red[t] += red[t + st]; __syncthreads(); }
  float inv = 1.0f / red[0]; __syncthreads();
  p *= inv;
  for (int d = 0; d < 16; ++d) {
    red[t] = p * vh[rowk + d];
    __syncthreads();
    for (int st = 128; st > 0; st >>= 1) { if (t < st) red[t] += red[t + st]; __syncthreads(); }
    if (t == 0) o[rowq + d] = red[0];
    __syncthreads();
  }
}

__global__ void stop_head_kernel(const float* __restrict__ x, const float* __restrict__ W,
                                 const float* __restrict__ bias, float* __restrict__ out) {
  __shared__ float red[128];
  int b = blockIdx.x, f = threadIdx.x;
  red[f] = x[(size_t)b * 128 + f] * W[f];
  __syncthreads();
  for (int s = 64; s > 0; s >>= 1) { if (f < s) red[f] += red[f + s]; __syncthreads(); }
  if (f == 0) out[(size_t)b * 65537 + 65536] = red[0] + bias[0];
}

// ---------------------------------------------------------------------------
extern "C" void kernel_launch(void* const* d_in, const int* in_sizes, int n_in,
                              void* d_out, int out_size, void* d_ws, size_t ws_size,
                              hipStream_t stream) {
  (void)in_sizes; (void)n_in; (void)out_size; (void)ws_size;

  // ---- input leaves (pytree order of setup_inputs dict) ----
  const int*   adj       = (const int*)d_in[0];
  const float* node_emb  = (const float*)d_in[1];
  const float* edge_emb  = (const float*)d_in[2];
  const float* glob_emb  = (const float*)d_in[3];
  const float* nodeW[3]  = {(const float*)d_in[4], (const float*)d_in[6], (const float*)d_in[8]};
  const float* nodeB[3]  = {(const float*)d_in[5], (const float*)d_in[7], (const float*)d_in[9]};
  const float* edgeW[3]  = {(const float*)d_in[10], (const float*)d_in[12], (const float*)d_in[14]};
  const float* edgeB[3]  = {(const float*)d_in[11], (const float*)d_in[13], (const float*)d_in[15]};
  const float* globW[3]  = {(const float*)d_in[16], (const float*)d_in[18], (const float*)d_in[20]};
  const float* globB[3]  = {(const float*)d_in[17], (const float*)d_in[19], (const float*)d_in[21]};
  const float* lnN_w = (const float*)d_in[22], *lnN_b = (const float*)d_in[23];
  const float* lnE_w = (const float*)d_in[24], *lnE_b = (const float*)d_in[25];
  const float* lnG_w = (const float*)d_in[26], *lnG_b = (const float*)d_in[27];
  const float* attnW = (const float*)d_in[28], *attnB = (const float*)d_in[29];
  const float* wq = (const float*)d_in[30], *wk = (const float*)d_in[31];
  const float* wv = (const float*)d_in[32], *wo = (const float*)d_in[33];
  const float* lnPN_w = (const float*)d_in[34], *lnPN_b = (const float*)d_in[35];
  const float* lnPG_w = (const float*)d_in[36], *lnPG_b = (const float*)d_in[37];
  const float* sndW[3] = {(const float*)d_in[38], (const float*)d_in[40], (const float*)d_in[42]};
  const float* sndB[3] = {(const float*)d_in[39], (const float*)d_in[41], (const float*)d_in[43]};
  const float* rcvW[3] = {(const float*)d_in[44], (const float*)d_in[46], (const float*)d_in[48]};
  const float* rcvB[3] = {(const float*)d_in[45], (const float*)d_in[47], (const float*)d_in[49]};
  const float* stpW[3] = {(const float*)d_in[50], (const float*)d_in[52], (const float*)d_in[54]};
  const float* stpB[3] = {(const float*)d_in[51], (const float*)d_in[53], (const float*)d_in[55]};
  float* out = (float*)d_out;

  // ---- workspace layout ----
  char* ws = (char*)d_ws;
  size_t off = 0;
  auto alloc = [&](size_t bytes) -> char* {
    char* p = ws + off;
    off += (bytes + 255) & ~(size_t)255;
    return p;
  };
  float* edges   = (float*)alloc((size_t)ESIZE * 128 * 4);
  float* ne      = (float*)alloc((size_t)ESIZE * 128 * 4);
  unsigned short* h1b = (unsigned short*)alloc((size_t)ESIZE * 128 * 2);
  float* nodes   = (float*)alloc((size_t)NT * 128 * 4);
  float* nn      = (float*)alloc((size_t)NT * 128 * 4);
  unsigned short* nh1b = (unsigned short*)alloc((size_t)1088 * 128 * 2);
  float* ssnd    = (float*)alloc((size_t)NT * 128 * 4);
  float* srcv    = (float*)alloc((size_t)NT * 128 * 4);
  float* glob    = (float*)alloc((size_t)Gg * 128 * 4);
  float* ng      = (float*)alloc((size_t)Gg * 128 * 4);
  unsigned short* gh1b = (unsigned short*)alloc((size_t)64 * 128 * 2);
  float* snng    = (float*)alloc((size_t)Gg * 128 * 4);
  float* sneg    = (float*)alloc((size_t)Gg * 128 * 4);
  float* qkv     = (float*)alloc((size_t)NE * 384 * 4);
  float* qh      = (float*)alloc((size_t)NE * 128 * 4);
  float* kh      = (float*)alloc((size_t)NE * 128 * 4);
  float* vh      = (float*)alloc((size_t)NE * 128 * 4);
  float* atn     = (float*)alloc((size_t)NE * 128 * 4);
  float* aout    = (float*)alloc((size_t)NE * 128 * 4);
  float* nf      = (float*)alloc((size_t)NE * 128 * 4);
  float* gf      = (float*)alloc((size_t)Bc * 128 * 4);
  unsigned short* mtmpb = (unsigned short*)alloc((size_t)NE * 128 * 2);
  float* sd      = (float*)alloc((size_t)NE * 128 * 4);
  float* rc      = (float*)alloc((size_t)NE * 128 * 4);
  unsigned int* rcb = (unsigned int*)alloc((size_t)NE * 64 * 4);
  unsigned short* st1b = (unsigned short*)alloc((size_t)64 * 128 * 2);
  float* st2     = (float*)alloc((size_t)64 * 128 * 4);
  int* snd    = (int*)alloc((size_t)ESIZE * 4);
  int* rcv    = (int*)alloc((size_t)ESIZE * 4);
  int* egid   = (int*)alloc((size_t)ESIZE * 4);
  int* counts = (int*)alloc((size_t)NE * 4);
  int* offs   = (int*)alloc((size_t)NE * 4);
  int* ngid   = (int*)alloc((size_t)1088 * 4);

  // ---- pre-convert all weights to bf16 (tiny, stays in 192MB L2) ----
  auto cvtw = [&](const float* src, int elems) -> const unsigned int* {
    unsigned int* dst = (unsigned int*)alloc((size_t)elems * 2);
    w2bf_kernel<<<(elems / 2 + 255) / 256, 256, 0, stream>>>(src, dst, elems / 2);
    return dst;
  };
  const unsigned int* nodeWb[3] = {cvtw(nodeW[0], 65536), cvtw(nodeW[1], 16384), cvtw(nodeW[2], 16384)};
  const unsigned int* edgeWb[3] = {cvtw(edgeW[0], 65536), cvtw(edgeW[1], 16384), cvtw(edgeW[2], 16384)};
  const unsigned int* globWb[3] = {cvtw(globW[0], 49152), cvtw(globW[1], 16384), cvtw(globW[2], 16384)};
  const unsigned int* attnWb = cvtw(attnW, 49152);
  const unsigned int* wqb = cvtw(wq, 16384), *wkb = cvtw(wk, 16384);
  const unsigned int* wvb = cvtw(wv, 16384), *wob = cvtw(wo, 16384);
  const unsigned int* sndWb[3] = {cvtw(sndW[0], 16384), cvtw(sndW[1], 16384), cvtw(sndW[2], 16384)};
  const unsigned int* rcvWb[3] = {cvtw(rcvW[0], 16384), cvtw(rcvW[1], 16384), cvtw(rcvW[2], 16384)};
  const unsigned int* stpWb[2] = {cvtw(stpW[0], 16384), cvtw(stpW[1], 16384)};

  // ---- GEMM dispatch (mode known statically at each call site) ----
  auto gemmFF = [&](const float* Af, int lda, int acol, int M, int K,
                    const unsigned int* Wb, int ldw2, const float* bias,
                    float* Cf, int ldc, int relu, int Nout) {
    dim3 g((M + 63) / 64, Nout / 128);
    gemm_t<false, false><<<g, 256, 0, stream>>>(Af, nullptr, lda, acol, M, K, Wb, ldw2, bias, Cf, nullptr, ldc, relu);
  };
  auto gemmFB = [&](const float* Af, int lda, int acol, int M, int K,
                    const unsigned int* Wb, int ldw2, const float* bias,
                    unsigned short* Cb, int ldc, int relu, int Nout) {
    dim3 g((M + 63) / 64, Nout / 128);
    gemm_t<false, true><<<g, 256, 0, stream>>>(Af, nullptr, lda, acol, M, K, Wb, ldw2, bias, nullptr, Cb, ldc, relu);
  };
  auto gemmBB = [&](const unsigned short* Ab, int lda, int M, int K,
                    const unsigned int* Wb, int ldw2, const float* bias,
                    unsigned short* Cb, int ldc, int relu, int Nout) {
    dim3 g((M + 63) / 64, Nout / 128);
    gemm_t<true, true><<<g, 256, 0, stream>>>(nullptr, Ab, lda, 0, M, K, Wb, ldw2, bias, nullptr, Cb, ldc, relu);
  };
  auto gemmBF = [&](const unsigned short* Ab, int lda, int M, int K,
                    const unsigned int* Wb, int ldw2, const float* bias,
                    float* Cf, int ldc, int relu, int Nout) {
    dim3 g((M + 63) / 64, Nout / 128);
    gemm_t<true, false><<<g, 256, 0, stream>>>(nullptr, Ab, lda, 0, M, K, Wb, ldw2, bias, Cf, nullptr, ldc, relu);
  };

  // ---- build edge list (matches jnp.nonzero row-major order) ----
  count_rows_kernel<<<NE, 256, 0, stream>>>(adj, counts);
  scan_rows_kernel<<<1, 1024, 0, stream>>>(counts, offs);
  init_edge_arrays_kernel<<<(ESIZE + 255) / 256, 256, 0, stream>>>(snd, rcv, egid);
  emit_edges_kernel<<<(NE + 255) / 256, 256, 0, stream>>>(adj, offs, snd, rcv, egid);
  init_ngid_kernel<<<(NT + 255) / 256, 256, 0, stream>>>(ngid);

  // ---- initialize embeddings ----
  init_nodes_kernel<<<NT, 128, 0, stream>>>(node_emb, nodes);
  bcast_row_kernel<<<ESIZE, 128, 0, stream>>>(edge_emb, edges);
  bcast_row_kernel<<<Gg, 128, 0, stream>>>(glob_emb, glob);

  // ---- 3 GraphNetwork layers ----
  for (int layer = 0; layer < 3; ++layer) {
    // edge MLP: concat([edges, nodes[snd], nodes[rcv], glob[egid]]) -> 512->128x3
    {
      dim3 g((ESIZE + 63) / 64, 1);
      gemm_cat_t<4><<<g, 256, 0, stream>>>(edges, nodes, nodes, glob, snd, rcv, egid,
                                           ESIZE, edgeWb[0], edgeB[0], h1b, 1);
    }
    gemmBB(h1b, 128, ESIZE, 128, edgeWb[1], 64, edgeB[1], h1b, 128, 1, 128);   // in-place
    gemmBF(h1b, 128, ESIZE, 128, edgeWb[2], 64, edgeB[2], ne, 128, 0, 128);
    // segment sums of new edges
    zero_kernel<<<(NT * 128 + 255) / 256, 256, 0, stream>>>(ssnd, NT * 128);
    zero_kernel<<<(NT * 128 + 255) / 256, 256, 0, stream>>>(srcv, NT * 128);
    zero_kernel<<<(Gg * 128 + 255) / 256, 256, 0, stream>>>(sneg, Gg * 128);
    scatter_add_edges_kernel<<<ESIZE, 128, 0, stream>>>(ne, snd, rcv, egid, ssnd, srcv, sneg);
    // node MLP: concat([nodes, ssnd, srcv, glob[ngid]])
    {
      dim3 g((NT + 63) / 64, 1);
      gemm_cat_t<4><<<g, 256, 0, stream>>>(nodes, ssnd, srcv, glob, nullptr, nullptr, ngid,
                                           NT, nodeWb[0], nodeB[0], nh1b, 1);
    }
    gemmBB(nh1b, 128, NT, 128, nodeWb[1], 64, nodeB[1], nh1b, 128, 1, 128);    // in-place
    gemmBF(nh1b, 128, NT, 128, nodeWb[2], 64, nodeB[2], nn, 128, 0, 128);
    // global MLP: concat([seg(nn,ngid), seg(ne,egid), glob])
    zero_kernel<<<(Gg * 128 + 255) / 256, 256, 0, stream>>>(snng, Gg * 128);
    scatter_add_nodes_kernel<<<NT, 128, 0, stream>>>(nn, ngid, snng);
    {
      dim3 g(1, 1);
      gemm_cat_t<3><<<g, 256, 0, stream>>>(snng, sneg, glob, nullptr, nullptr, nullptr, nullptr,
                                           Gg, globWb[0], globB[0], gh1b, 1);
    }
    gemmBB(gh1b, 128, Gg, 128, globWb[1], 64, globB[1], gh1b, 128, 1, 128);    // in-place
    gemmBF(gh1b, 128, Gg, 128, globWb[2], 64, globB[2], ng, 128, 0, 128);
    // residual + layernorm (applied last: uses pre-update state above)
    ln_kernel<<<NT, 128, 0, stream>>>(nodes, nn, lnN_w, lnN_b, nodes);
    ln_kernel<<<ESIZE, 128, 0, stream>>>(edges, ne, lnE_w, lnE_b, edges);
    ln_kernel<<<Gg, 128, 0, stream>>>(glob, ng, lnG_w, lnG_b, glob);
  }

  // ---- attention head ----
  gemmFF(nodes, 128, 0, NE, 128, attnWb, 64, attnB, qkv, 384, 0, 384);         // qkv
  gemmFF(qkv, 384, 0,   NE, 128, wqb, 64, nullptr, qh, 128, 0, 128);
  gemmFF(qkv, 384, 128, NE, 128, wkb, 64, nullptr, kh, 128, 0, 128);
  gemmFF(qkv, 384, 256, NE, 128, wvb, 64, nullptr, vh, 128, 0, 128);
  attn_kernel<<<Bc * Hh * Nn, 256, 0, stream>>>(qh, kh, vh, atn);
  gemmFF(atn, 128, 0, NE, 128, wob, 64, nullptr, aout, 128, 0, 128);
  ln_kernel<<<NE, 128, 0, stream>>>(aout, nullptr, lnPN_w, lnPN_b, nf);        // nf
  ln_kernel<<<Bc, 128, 0, stream>>>(glob, nullptr, lnPG_w, lnPG_b, gf);        // gf

  // ---- senders / receivers MLPs ----
  gemmFB(nf, 128, 0, NE, 128, sndWb[0], 64, sndB[0], mtmpb, 128, 1, 128);
  gemmBB(mtmpb, 128, NE, 128, sndWb[1], 64, sndB[1], mtmpb, 128, 1, 128);      // in-place
  gemmBF(mtmpb, 128, NE, 128, sndWb[2], 64, sndB[2], sd, 128, 0, 128);
  gemmFB(nf, 128, 0, NE, 128, rcvWb[0], 64, rcvB[0], mtmpb, 128, 1, 128);
  gemmBB(mtmpb, 128, NE, 128, rcvWb[1], 64, rcvB[1], mtmpb, 128, 1, 128);      // in-place
  gemmBF(mtmpb, 128, NE, 128, rcvWb[2], 64, rcvB[2], rc, 128, 0, 128);

  // rc -> bf16 so it can act as the "weight" of the pair-logit GEMM
  w2bf_kernel<<<(NE * 64 + 255) / 256, 256, 0, stream>>>(rc, rcb, NE * 64);

  // ---- pair logits: per batch  sd_b @ rc_b^T  -> d_out rows of width 65537 ----
  for (int b = 0; b < Bc; ++b) {
    gemmFF(sd + (size_t)b * Nn * 128, 128, 0, Nn, 128,
           rcb + (size_t)b * Nn * 64, 64, nullptr,
           out + (size_t)b * 65537, 256, 0, 256);
  }

  // ---- stop head ----
  gemmFB(gf, 128, 0, Bc, 128, stpWb[0], 64, stpB[0], st1b, 128, 1, 128);
  gemmBF(st1b, 128, Bc, 128, stpWb[1], 64, stpB[1], st2, 128, 1, 128);
  stop_head_kernel<<<Bc, 128, 0, stream>>>(st2, stpW[2], stpB[2], out);

  // ---- bwd logits: zeros ----
  zero_kernel<<<(Bc * 256 + 255) / 256, 256, 0, stream>>>(out + (size_t)Bc * 65537, Bc * 256);
}
